// Agent_65996467470875
// MI455X (gfx1250) — matrix operand order repaired and checked by reading, not compiled, not run on hardware
//
#include <hip/hip_runtime.h>

// Problem dims (hard-coded in the reference)
#define T_DIM 1024
#define B_DIM 512
#define OBS_DIM 128
#define H_DIM 64
#define G_DIM 192            // 3*H
#define A_DIM 18
#define M_TOT (T_DIM * B_DIM)

typedef __bf16 bf16;
typedef __attribute__((ext_vector_type(16))) __bf16 v16bf;
typedef __attribute__((ext_vector_type(8)))  __bf16 v8bf;
typedef __attribute__((ext_vector_type(4)))  __bf16 v4bf;
typedef __attribute__((ext_vector_type(8)))  float  v8f;

// gi workspace layout: [t][bg(32)][gate(3)][col(64)][m(16)]  (bf16)
#define GI_T_STRIDE (32 * 3 * 64 * 16)   // 98304
// ys workspace layout: [t][bg(32)][col(64)][m(16)]           (bf16)
#define YS_T_STRIDE (32 * 64 * 16)       // 32768

__device__ __forceinline__ v8f wmma_bf16(v16bf a, v16bf b, v8f c) {
  return __builtin_amdgcn_wmma_f32_16x16x32_bf16(false, a, false, b, (short)0, c,
                                                 false, false);
}
__device__ __forceinline__ float sigmoidf(float x) {
  return 1.0f / (1.0f + __expf(-x));
}
__device__ __forceinline__ float fast_tanhf(float x) {
  // tanh(x) = 2*sigmoid(2x) - 1, on the v_exp_f32 transcendental path
  return 2.0f / (1.0f + __expf(-2.0f * x)) - 1.0f;
}

// A-fragment inverse mapping (16x32 bf16 tile): element (m, kk) lives at
// lane = m + 16*halfA, slot eA, with:
__device__ __forceinline__ int a_half(int kk) { return kk >> 4; }
__device__ __forceinline__ int a_slot(int kk) { return kk & 15; }

// ---- Kernel 1: gi = relu(x@W1 + b1) @ Wi + bi  (parallel over T*B) --------
__global__ __launch_bounds__(256) void stem_gi_kernel(
    const float* __restrict__ x, const float* __restrict__ W1,
    const float* __restrict__ b1, const float* __restrict__ Wi,
    const float* __restrict__ bi, bf16* __restrict__ gi) {
  __shared__ v16bf s_xfrag[8 * 4 * 32];    // x A-frags  [mt][kt][lane]  32KB
  __shared__ v16bf s_w1frag[4 * 4 * 32];   // W1 B-frags [kt][nt][lane]  16KB
  __shared__ v16bf s_wifrag[2 * 12 * 32];  // Wi B-frags [kt][nt][lane]  24KB
  __shared__ v16bf s_zfrag[8 * 2 * 32];    // z A-frags  [mt][kt][lane]  16KB

  const int tid = threadIdx.x;
  const int lane = tid & 31;
  const int wv = tid >> 5;              // 8 waves; wave = 16-row M-tile
  const int half = lane >> 4;
  const int nn = lane & 15;
  const long long row0 = (long long)blockIdx.x * 128;   // 128 rows of [T*B]
  const int tt = (int)(row0 >> 9);                      // t index
  const int bg0 = ((int)row0 & 511) >> 4;               // first 16-row b-group

  // ---- pack x tile [128][128] into A-fragment layout (coalesced float4) ----
  {
    const float4* x4 = (const float4*)(x + row0 * OBS_DIM);
    for (int i = tid; i < 128 * 32; i += 256) {
      const int r = i >> 5;
      const int c = (i & 31) * 4;       // 4-aligned => 4 consecutive slots
      const float4 v = x4[i];
      const int mt = r >> 4, m = r & 15;
      const int kt = c >> 5, kk = c & 31;
      const int laneA = m + (a_half(kk) << 4);
      bf16* dst = (bf16*)&s_xfrag[(mt * 4 + kt) * 32 + laneA] + a_slot(kk);
      v4bf pv;
      pv[0] = (bf16)v.x; pv[1] = (bf16)v.y; pv[2] = (bf16)v.z; pv[3] = (bf16)v.w;
      *(v4bf*)dst = pv;
    }
  }
  // ---- pack W1 [128][64] B-fragments (one (frag,lane) pair per iteration) --
  for (int p = tid; p < 16 * 32; p += 256) {
    const int frag = p >> 5, ln = p & 31;
    const int kt = frag >> 2, nt = frag & 3;
    const int hb = ln >> 4, n = ln & 15;
    v16bf f;
#pragma unroll
    for (int e = 0; e < 16; ++e)
      f[e] = (bf16)W1[(kt * 32 + e + 16 * hb) * H_DIM + nt * 16 + n];
    s_w1frag[frag * 32 + ln] = f;
  }
  // ---- pack Wi [64][192] B-fragments ----
  for (int p = tid; p < 24 * 32; p += 256) {
    const int frag = p >> 5, ln = p & 31;
    const int kt = frag / 12, nt = frag % 12;
    const int hb = ln >> 4, n = ln & 15;
    v16bf f;
#pragma unroll
    for (int e = 0; e < 16; ++e)
      f[e] = (bf16)Wi[(kt * 32 + e + 16 * hb) * G_DIM + nt * 16 + n];
    s_wifrag[frag * 32 + ln] = f;
  }
  __syncthreads();

  // ---- z = relu(x@W1 + b1) ----
  v16bf ax[4];
#pragma unroll
  for (int kt = 0; kt < 4; ++kt) ax[kt] = s_xfrag[(wv * 4 + kt) * 32 + lane];
#pragma unroll
  for (int nt = 0; nt < 4; ++nt) {
    v8f acc = {};
#pragma unroll
    for (int kt = 0; kt < 4; ++kt)
      acc = wmma_bf16(ax[kt], s_w1frag[(kt * 4 + nt) * 32 + lane], acc);
    const int col = nt * 16 + nn;
    const float bias = b1[col];
    const int ktz = col >> 5, kk = col & 31;
    const int ha = a_half(kk), ea = a_slot(kk);
#pragma unroll
    for (int e = 0; e < 8; ++e) {
      const int m = 8 * half + e;
      float v = acc[e] + bias;
      v = v > 0.f ? v : 0.f;
      ((bf16*)&s_zfrag[(wv * 2 + ktz) * 32 + m + 16 * ha])[ea] = (bf16)v;
    }
  }
  // z tiles are wave-private (written & read by same wave): no barrier needed.

  // ---- gi = z@Wi + bi, stored in consumption order for the scan kernel ----
  v16bf az[2] = {s_zfrag[(wv * 2) * 32 + lane], s_zfrag[(wv * 2 + 1) * 32 + lane]};
  const int bg = bg0 + wv;
#pragma unroll
  for (int nt = 0; nt < 12; ++nt) {
    v8f acc = {};
#pragma unroll
    for (int kt = 0; kt < 2; ++kt)
      acc = wmma_bf16(az[kt], s_wifrag[(kt * 12 + nt) * 32 + lane], acc);
    const int col = nt * 16 + nn;
    const float bias = bi[col];
    const int g = col >> 6, j = col & 63;
    v8bf o;
#pragma unroll
    for (int e = 0; e < 8; ++e) o[e] = (bf16)(acc[e] + bias);
    bf16* dst = gi + ((size_t)tt * GI_T_STRIDE) +
                (((size_t)bg * 3 + g) * 64 + j) * 16 + 8 * half;
    *(v8bf*)dst = o;    // 16B store, coalesced across lanes
  }
}

// ---- Kernel 2: GRU scan. Batch rows independent -> 32 blocks, no global
//      sync. Ping-pong h state => ONE barrier per step; mask folded into the
//      h update; gi prefetched ahead of the dependency chain.
__global__ __launch_bounds__(128) void gru_scan_kernel(
    const unsigned char* __restrict__ mask,   // jnp.bool_: 1 byte/elt
    const float* __restrict__ carry, const float* __restrict__ Wh,
    const float* __restrict__ bhn, const bf16* __restrict__ gi,
    bf16* __restrict__ ys, float* __restrict__ h_final) {
  __shared__ v16bf s_hfrag[2][2 * 32];      // ping-pong packed A-frags of h
  __shared__ float s_h32[2][64 * 16];       // ping-pong h f32, [col][m]

  const int tid = threadIdx.x;
  const int lane = tid & 31;
  const int wv = tid >> 5;                  // wave w owns gate cols [16w,16w+16)
  const int half = lane >> 4;
  const int nn = lane & 15;
  const int bg = blockIdx.x;
  const int b0 = bg * 16;

  // ---- init h_in(0) = initial_carry into buffer 0 ----
  {
    const int r = tid >> 3;
    const int c0 = (tid & 7) * 8;
    const float4 v0 = *(const float4*)(carry + (b0 + r) * H_DIM + c0);
    const float4 v1 = *(const float4*)(carry + (b0 + r) * H_DIM + c0 + 4);
    const float vv[8] = {v0.x, v0.y, v0.z, v0.w, v1.x, v1.y, v1.z, v1.w};
    v8bf pv;
#pragma unroll
    for (int cc = 0; cc < 8; ++cc) {
      s_h32[0][(c0 + cc) * 16 + r] = vv[cc];
      pv[cc] = (bf16)vv[cc];
    }
    const int kt = c0 >> 5, kk = c0 & 31;
    bf16* dst = (bf16*)&s_hfrag[0][kt * 32 + r + 16 * a_half(kk)] + a_slot(kk);
    *(v8bf*)dst = pv;   // c0 is 8-aligned => 8 contiguous slots
  }

  // ---- Wh B-fragments straight from global into VGPRs (kept for all steps) -
  v16bf bw[3][2];
#pragma unroll
  for (int g = 0; g < 3; ++g)
#pragma unroll
    for (int kt = 0; kt < 2; ++kt) {
      v16bf f;
#pragma unroll
      for (int e = 0; e < 16; ++e)
        f[e] = (bf16)Wh[(kt * 32 + e + 16 * half) * G_DIM + g * 64 + wv * 16 + nn];
      bw[g][kt] = f;
    }

  const int col = wv * 16 + nn;             // hidden/gate column 0..63
  const float bhn_c = bhn[col];
  // per-lane initial_carry values for this lane's 8 (m, col) cells
  float carry_reg[8];
#pragma unroll
  for (int e = 0; e < 8; ++e)
    carry_reg[e] = carry[(b0 + 8 * half + e) * H_DIM + col];

  // fixed packed-fragment destination for this lane's column
  const int ktc = col >> 5, kkc = col & 31;
  const int hac = a_half(kkc), eac = a_slot(kkc);

  const bf16* gbase = gi + (((size_t)bg * 3) * 64 + col) * 16 + 8 * half;
  bf16* ybase = ys + ((size_t)bg * 64 + col) * 16 + 8 * half;
  const unsigned char* mbase = mask + b0 + 8 * half;

  for (int t = 0; t < T_DIM; ++t) {
    const int pb = t & 1;        // read buffer
    const int wb = pb ^ 1;       // write buffer

    // prefetch gi for t+4 (independent of the h dependency chain)
    if (t + 4 < T_DIM) {
      const bf16* pf = gbase + (size_t)(t + 4) * GI_T_STRIDE;
      __builtin_prefetch(pf, 0, 1);
      __builtin_prefetch(pf + 64 * 16, 0, 1);
      __builtin_prefetch(pf + 2 * 64 * 16, 0, 1);
    }

    __syncthreads();   // h_in(t) writes (from step t-1) visible

    const v16bf a0 = s_hfrag[pb][lane];
    const v16bf a1 = s_hfrag[pb][32 + lane];
    v8f ar = {}, au = {}, an = {};
    ar = wmma_bf16(a0, bw[0][0], ar);
    au = wmma_bf16(a0, bw[1][0], au);
    an = wmma_bf16(a0, bw[2][0], an);
    ar = wmma_bf16(a1, bw[0][1], ar);
    au = wmma_bf16(a1, bw[1][1], au);
    an = wmma_bf16(a1, bw[2][1], an);

    const bf16* gp = gbase + (size_t)t * GI_T_STRIDE;
    const v8bf gr = *(const v8bf*)(gp);
    const v8bf gu = *(const v8bf*)(gp + 64 * 16);
    const v8bf gn = *(const v8bf*)(gp + 2 * 64 * 16);

    // old h for this lane's 8 cells (contiguous: [col][m] layout)
    const float4 ho0 = *(const float4*)&s_h32[pb][col * 16 + 8 * half];
    const float4 ho1 = *(const float4*)&s_h32[pb][col * 16 + 8 * half + 4];
    const float hof[8] = {ho0.x, ho0.y, ho0.z, ho0.w, ho1.x, ho1.y, ho1.z, ho1.w};

    // mask for step t+1 (one 8-byte load; folded into the h update below)
    unsigned long long mv = 0;
    if (t + 1 < T_DIM)
      mv = *(const unsigned long long*)(mbase + (size_t)(t + 1) * B_DIM);

    float hnext[8];
    v8bf ybuf;
#pragma unroll
    for (int e = 0; e < 8; ++e) {
      const float rg = sigmoidf((float)gr[e] + ar[e]);
      const float ug = sigmoidf((float)gu[e] + au[e]);
      const float ng = fast_tanhf((float)gn[e] + rg * (an[e] + bhn_c));
      const float hn = (1.0f - ug) * ng + ug * hof[e];
      ybuf[e] = (bf16)hn;
      const bool rst = ((mv >> (8 * e)) & 0xff) != 0;
      hnext[e] = rst ? carry_reg[e] : hn;
    }
    *(v8bf*)(ybase + (size_t)t * YS_T_STRIDE) = ybuf;   // h_out(t)

    // write h_in(t+1) into the other buffer (no extra barrier needed:
    // step t+1's top barrier orders these writes against its reads)
    float4* hp32 = (float4*)&s_h32[wb][col * 16 + 8 * half];
    hp32[0] = make_float4(hnext[0], hnext[1], hnext[2], hnext[3]);
    hp32[1] = make_float4(hnext[4], hnext[5], hnext[6], hnext[7]);
#pragma unroll
    for (int e = 0; e < 8; ++e)
      ((bf16*)&s_hfrag[wb][ktc * 32 + (8 * half + e) + 16 * hac])[eac] =
          (bf16)hnext[e];
  }
  __syncthreads();

  // h_final = h_out(T-1): T even => it lives in buffer 0 (mask beyond last
  // step doesn't exist, so hnext == h_out for the final write)
  for (int i = tid; i < 16 * 64; i += 128) {
    const int r = i >> 6, c = i & 63;
    h_final[(b0 + r) * H_DIM + c] = s_h32[T_DIM & 1][c * 16 + r];
  }
}

// ---- Kernel 3: y = relu(ys@W2+b2); [logits|value] = y @ [Wl|Wv] + [bl|bv] --
__global__ __launch_bounds__(256) void head_kernel(
    const bf16* __restrict__ ys, const float* __restrict__ W2,
    const float* __restrict__ b2, const float* __restrict__ Wl,
    const float* __restrict__ bl, const float* __restrict__ Wv,
    const float* __restrict__ bv, float* __restrict__ logits,
    float* __restrict__ value) {
  __shared__ v16bf s_infrag[8 * 2 * 32];   // ys A-frags   16KB
  __shared__ v16bf s_w2frag[2 * 4 * 32];   // W2 B-frags    8KB
  __shared__ v16bf s_yfrag[8 * 2 * 32];    // y  A-frags   16KB
  __shared__ v16bf s_wlvfrag[2 * 2 * 32];  // [Wl|Wv|0] B-frags  4KB

  const int tid = threadIdx.x;
  const int lane = tid & 31;
  const int wv = tid >> 5;
  const int half = lane >> 4;
  const int nn = lane & 15;
  const long long row0 = (long long)blockIdx.x * 128;
  const int tt = (int)(row0 >> 9);
  const int bg0 = ((int)row0 & 511) >> 4;

  // ---- pack ys tile from [t][bg][col][m] layout (contiguous 16B reads) ----
  for (int p = tid; p < 8 * 64 * 2; p += 256) {
    const int bgl = p >> 7;              // 0..7 (= m-tile)
    const int c = (p >> 1) & 63;
    const int m0 = (p & 1) * 8;
    const v8bf val = *(const v8bf*)(ys + ((size_t)tt * YS_T_STRIDE) +
                                    (((size_t)(bg0 + bgl)) * 64 + c) * 16 + m0);
    const int kt = c >> 5, kk = c & 31;
    const int ha = a_half(kk), ea = a_slot(kk);
#pragma unroll
    for (int e2 = 0; e2 < 8; ++e2)
      ((bf16*)&s_infrag[(bgl * 2 + kt) * 32 + (m0 + e2) + 16 * ha])[ea] = val[e2];
  }
  // ---- pack W2 [64][64] B-fragments ----
  for (int p = tid; p < 8 * 32; p += 256) {
    const int frag = p >> 5, ln = p & 31;
    const int kt = frag >> 2, nt = frag & 3;
    const int hb = ln >> 4, n = ln & 15;
    v16bf f;
#pragma unroll
    for (int e = 0; e < 16; ++e)
      f[e] = (bf16)W2[(kt * 32 + e + 16 * hb) * H_DIM + nt * 16 + n];
    s_w2frag[frag * 32 + ln] = f;
  }
  // ---- pack [Wl | Wv | 0] padded [64][32] B-fragments ----
  for (int p = tid; p < 4 * 32; p += 256) {
    const int frag = p >> 5, ln = p & 31;
    const int kt = frag >> 1, nt = frag & 1;
    const int hb = ln >> 4, n = ln & 15;
    const int c = nt * 16 + n;
    v16bf f;
#pragma unroll
    for (int e = 0; e < 16; ++e) {
      const int k = kt * 32 + e + 16 * hb;
      const float w = (c < A_DIM) ? Wl[k * A_DIM + c] : (c == A_DIM ? Wv[k] : 0.f);
      f[e] = (bf16)w;
    }
    s_wlvfrag[frag * 32 + ln] = f;
  }
  __syncthreads();

  // ---- y = relu(ys@W2 + b2) ----
  v16bf ai[2] = {s_infrag[(wv * 2) * 32 + lane], s_infrag[(wv * 2 + 1) * 32 + lane]};
#pragma unroll
  for (int nt = 0; nt < 4; ++nt) {
    v8f acc = {};
#pragma unroll
    for (int kt = 0; kt < 2; ++kt)
      acc = wmma_bf16(ai[kt], s_w2frag[(kt * 4 + nt) * 32 + lane], acc);
    const int c = nt * 16 + nn;
    const float bias = b2[c];
    const int ktz = c >> 5, kk = c & 31;
    const int ha = a_half(kk), ea = a_slot(kk);
#pragma unroll
    for (int e = 0; e < 8; ++e) {
      float v = acc[e] + bias;
      v = v > 0.f ? v : 0.f;
      ((bf16*)&s_yfrag[(wv * 2 + ktz) * 32 + (8 * half + e) + 16 * ha])[ea] = (bf16)v;
    }
  }
  // y tiles are wave-private: no barrier needed.

  // ---- [logits | value] ----
  v16bf ay[2] = {s_yfrag[(wv * 2) * 32 + lane], s_yfrag[(wv * 2 + 1) * 32 + lane]};
#pragma unroll
  for (int nt = 0; nt < 2; ++nt) {
    v8f acc = {};
#pragma unroll
    for (int kt = 0; kt < 2; ++kt)
      acc = wmma_bf16(ay[kt], s_wlvfrag[(kt * 2 + nt) * 32 + lane], acc);
    const int c = nt * 16 + nn;
    const float bias = (c < A_DIM) ? bl[c] : (c == A_DIM ? bv[0] : 0.f);
#pragma unroll
    for (int e = 0; e < 8; ++e) {
      const long long m = row0 + wv * 16 + 8 * half + e;
      const float v = acc[e] + bias;
      if (c < A_DIM)       logits[m * A_DIM + c] = v;
      else if (c == A_DIM) value[m] = v;
    }
  }
}

// ---------------------------------------------------------------------------
extern "C" void kernel_launch(void* const* d_in, const int* in_sizes, int n_in,
                              void* d_out, int out_size, void* d_ws,
                              size_t ws_size, hipStream_t stream) {
  const float* x            = (const float*)d_in[0];
  const unsigned char* mask = (const unsigned char*)d_in[1];
  const float* carry        = (const float*)d_in[2];
  const float* W1           = (const float*)d_in[3];
  const float* b1           = (const float*)d_in[4];
  const float* Wi           = (const float*)d_in[5];
  const float* bi           = (const float*)d_in[6];
  const float* Wh           = (const float*)d_in[7];
  const float* bhn          = (const float*)d_in[8];
  const float* W2           = (const float*)d_in[9];
  const float* b2           = (const float*)d_in[10];
  const float* Wl           = (const float*)d_in[11];
  const float* bl           = (const float*)d_in[12];
  const float* Wv           = (const float*)d_in[13];
  const float* bv           = (const float*)d_in[14];

  float* out     = (float*)d_out;
  float* h_final = out;                               // [B, H]
  float* logits  = out + B_DIM * H_DIM;               // [T, B, A]
  float* value   = logits + (size_t)M_TOT * A_DIM;    // [T, B]

  bf16* gi = (bf16*)d_ws;                             // [T][32][3][64][16] bf16
  bf16* ys = gi + (size_t)T_DIM * GI_T_STRIDE;        // [T][32][64][16]    bf16

  stem_gi_kernel<<<M_TOT / 128, 256, 0, stream>>>(x, W1, b1, Wi, bi, gi);
  gru_scan_kernel<<<B_DIM / 16, 128, 0, stream>>>(mask, carry, Wh, bhn, gi, ys,
                                                  h_final);
  head_kernel<<<M_TOT / 128, 256, 0, stream>>>(ys, W2, b2, Wl, bl, Wv, bv,
                                               logits, value);
}
